// GlobalMetaAggregator_38671885533801
// MI455X (gfx1250) — matrix operand order repaired and checked by reading, not compiled, not run on hardware
//
#include <hip/hip_runtime.h>
#include <hip/hip_bf16.h>

// ---------------------------------------------------------------------------
// GlobalMetaAggregator on MI455X (gfx1250).
//  * All GEMMs through v_wmma_f32_16x16x32_f16 (wave32, f32 accum).
//  * One up-front pass converts fp32 activations -> f16 and weights ->
//    transposed (n-major) f16, so every GEMM inner loop is a guard-free,
//    conversion-free, DOUBLE-BUFFERED async pipeline using CDNA5
//    GLOBAL_LOAD_ASYNC_TO_LDS_B128 (+ s_wait_asynccnt).
// ---------------------------------------------------------------------------

typedef __attribute__((ext_vector_type(16))) _Float16 v16h;
typedef __attribute__((ext_vector_type(8)))  _Float16 v8h;
typedef __attribute__((ext_vector_type(8)))  float    v8f;

#define LDS_STRIDE 40                      // 32 halves + 8 pad -> 80B rows, 16B aligned
#define TILE_BYTES (128 * LDS_STRIDE * 2)  // one LDS tile buffer

__device__ __forceinline__ v16h make_frag(const _Float16* p0, const _Float16* p1) {
    v8h lo = *(const v8h*)p0;
    v8h hi = *(const v8h*)p1;
    return __builtin_shufflevector(lo, hi, 0,1,2,3,4,5,6,7,8,9,10,11,12,13,14,15);
}

// CDNA5 async global->LDS 16-byte copy (ASYNCcnt). lds_off = LDS byte address
// (low 32 bits of the generic pointer per the LDS aperture mapping).
__device__ __forceinline__ void async_b128(uint32_t lds_off, const _Float16* g) {
    asm volatile("global_load_async_to_lds_b128 %0, %1, off"
                 :: "v"(lds_off), "v"(g)
                 : "memory");
}
__device__ __forceinline__ void wait_async0() {
    asm volatile("s_wait_asynccnt 0x0" ::: "memory");
}

// ---------------------------------------------------------------------------
// WMMA GEMM: C[M,Nn] = A[M,K] * Bt[n][k] (+bias[col]) (+PReLU)
//   A : f16 row-major, leading dim lda          (M % 128 == 0)
//   Bt: f16, PRE-TRANSPOSED [NnPad>=128-tile rows, K], guard-free
//   C : TO (f16 or f32), leading dim ldc        (cols guarded by Nn)
// Block 256 threads (8 waves, 4x2). Tile 128x128x32, double-buffered async.
// ---------------------------------------------------------------------------
template <typename TO>
__global__ __launch_bounds__(256)
void gemm_wmma_async(const _Float16* __restrict__ A, int lda,
                     const _Float16* __restrict__ Bt,
                     TO* __restrict__ C, int ldc,
                     int M, int Nn, int K,
                     const float* __restrict__ bias,
                     const float* __restrict__ prelu_alpha) {
    __shared__ _Float16 sA[2][128 * LDS_STRIDE];
    __shared__ _Float16 sB[2][128 * LDS_STRIDE];

    const int tid  = threadIdx.x;
    const int lane = tid & 31;
    const int wid  = tid >> 5;
    const int wm0  = (wid & 3) * 32;
    const int wn0  = (wid >> 2) * 64;
    const int m0   = blockIdx.y * 128;
    const int n0   = blockIdx.x * 128;

    // Per-thread async chunk assignment: tile = 512 x 16B chunks; 2 per thread.
    // chunk id c: row = c>>2, 16B-chunk col = c&3.
    const int c0 = tid, c1 = tid + 256;
    const int r0 = c0 >> 2, r1 = c1 >> 2;
    const int h0 = (c0 & 3) * 8, h1 = (c1 & 3) * 8;   // col offset in halves

    const _Float16* gA0 = A  + (size_t)(m0 + r0) * lda + h0;
    const _Float16* gA1 = A  + (size_t)(m0 + r1) * lda + h1;
    const _Float16* gB0 = Bt + (size_t)(n0 + r0) * K   + h0;
    const _Float16* gB1 = Bt + (size_t)(n0 + r1) * K   + h1;

    const uint32_t sAb = (uint32_t)(uintptr_t)&sA[0][0];
    const uint32_t sBb = (uint32_t)(uintptr_t)&sB[0][0];
    const uint32_t oA0 = (uint32_t)(r0 * (LDS_STRIDE * 2) + (c0 & 3) * 16);
    const uint32_t oA1 = (uint32_t)(r1 * (LDS_STRIDE * 2) + (c1 & 3) * 16);
    const uint32_t oB0 = oA0, oB1 = oA1;

    v8f acc[2][4] = {};

    // prologue: stage k-tile 0 into buffer 0
    async_b128(sAb + oA0, gA0);
    async_b128(sAb + oA1, gA1);
    async_b128(sBb + oB0, gB0);
    async_b128(sBb + oB1, gB1);

    const int T = K >> 5;
    for (int t = 0; t < T; ++t) {
        const int buf = t & 1;
        wait_async0();        // my async stores for buf are done
        __syncthreads();      // everyone's are; everyone done reading buf^1

        if (t + 1 < T) {      // stage next k-tile into the freed buffer
            const uint32_t bo = (uint32_t)(buf ^ 1) * (uint32_t)TILE_BYTES;
            const int kh = (t + 1) << 5;   // k offset in halves
            async_b128(sAb + bo + oA0, gA0 + kh);
            async_b128(sAb + bo + oA1, gA1 + kh);
            async_b128(sBb + bo + oB0, gB0 + kh);
            async_b128(sBb + bo + oB1, gB1 + kh);
        }

        // ---- WMMA fragments (ISA 7.12.2 layouts) ----
        const int rl = lane & 15;
        const int ah = (lane >> 4) * 8;    // A: K-half base 0 / 8 (+16 hi chunk)
        const int bh = (lane >> 4) * 8;    // B: K-half base 0 / 8 (+16 hi chunk)
        v16h afr[2];
#pragma unroll
        for (int wm = 0; wm < 2; ++wm) {
            const _Float16* p = &sA[buf][(wm0 + wm * 16 + rl) * LDS_STRIDE];
            afr[wm] = make_frag(p + ah, p + ah + 16);
        }
        v16h bfr[4];
#pragma unroll
        for (int wn = 0; wn < 4; ++wn) {
            const _Float16* p = &sB[buf][(wn0 + wn * 16 + rl) * LDS_STRIDE];
            bfr[wn] = make_frag(p + bh, p + bh + 16);
        }
#pragma unroll
        for (int wm = 0; wm < 2; ++wm)
#pragma unroll
            for (int wn = 0; wn < 4; ++wn)
                acc[wm][wn] = __builtin_amdgcn_wmma_f32_16x16x32_f16(
                    false, afr[wm], false, bfr[wn], (short)0, acc[wm][wn],
                    false, false);
    }

    // ---- epilogue: C/D layout: row = j + 8*(lane>>4), col = lane&15 ----
    const float pa = prelu_alpha ? *prelu_alpha : 0.0f;
    const bool has_prelu = (prelu_alpha != nullptr);
#pragma unroll
    for (int wm = 0; wm < 2; ++wm) {
#pragma unroll
        for (int wn = 0; wn < 4; ++wn) {
            const int col = n0 + wn0 + wn * 16 + (lane & 15);
            if (col < Nn) {
                const float bb = bias ? bias[col] : 0.0f;
                const int rbase = m0 + wm0 + wm * 16 + ((lane >> 4) << 3);
#pragma unroll
                for (int j = 0; j < 8; ++j) {
                    float v = acc[wm][wn][j] + bb;
                    if (has_prelu) v = (v >= 0.0f) ? v : pa * v;
                    C[(size_t)(rbase + j) * ldc + col] = (TO)v;
                }
            }
        }
    }
}

// ---------------------------------------------------------------------------
// fp32 -> f16 elementwise convert (8 elems/thread, vectorized).
// ---------------------------------------------------------------------------
__global__ __launch_bounds__(256)
void cvt_f16_kernel(const float* __restrict__ src, _Float16* __restrict__ dst,
                    size_t n) {
    size_t i = ((size_t)blockIdx.x * 256 + threadIdx.x) * 8;
    if (i + 8 <= n) {
        float4 a = *(const float4*)(src + i);
        float4 b = *(const float4*)(src + i + 4);
        v8h o;
        o[0] = (_Float16)a.x; o[1] = (_Float16)a.y;
        o[2] = (_Float16)a.z; o[3] = (_Float16)a.w;
        o[4] = (_Float16)b.x; o[5] = (_Float16)b.y;
        o[6] = (_Float16)b.z; o[7] = (_Float16)b.w;
        *(v8h*)(dst + i) = o;
    }
}

// ---------------------------------------------------------------------------
// Weight transpose+convert: dst[b][n][k] = (f16)src[b][k][n]; rows n>=Nn zero.
// grid (K/32, NnPad/32, batch), 256 threads, LDS 32x33 tile.
// ---------------------------------------------------------------------------
__global__ __launch_bounds__(256)
void transpose_cvt_kernel(const float* __restrict__ src, _Float16* __restrict__ dst,
                          int K, int Nn, size_t src_bs, size_t dst_bs) {
    __shared__ float tile[32][33];
    src += blockIdx.z * src_bs;
    dst += blockIdx.z * dst_bs;
    const int k0 = blockIdx.x * 32, n0 = blockIdx.y * 32;
    const int tx = threadIdx.x & 31, ty = threadIdx.x >> 5;
#pragma unroll
    for (int i = 0; i < 4; ++i) {
        const int k = k0 + ty + 8 * i, n = n0 + tx;
        tile[ty + 8 * i][tx] = (n < Nn) ? src[(size_t)k * Nn + n] : 0.0f;
    }
    __syncthreads();
#pragma unroll
    for (int i = 0; i < 4; ++i) {
        const int n = n0 + ty + 8 * i;
        dst[(size_t)n * K + k0 + tx] = (_Float16)tile[tx][ty + 8 * i];
    }
}

// ---------------------------------------------------------------------------
// Sigmoid-softmax attention pool: f[N, Cdim, D] (f16) x w[D] -> out[N, D] f16.
// One wave per node; 8 nodes per block.
// ---------------------------------------------------------------------------
__global__ __launch_bounds__(256)
void attn_pool_kernel(const _Float16* __restrict__ f, const float* __restrict__ w,
                      _Float16* __restrict__ out,
                      int Nn, int Cdim, int D, int f_nstride, int out_nstride) {
    const int lane = threadIdx.x & 31;
    const int wid  = threadIdx.x >> 5;
    const int node = blockIdx.x * 8 + wid;
    if (node >= Nn) return;
    const _Float16* fp = f + (size_t)node * f_nstride;
    const int per = D >> 5;

    float s[8];
    for (int c = 0; c < Cdim; ++c) {
        float a = 0.0f;
        const _Float16* row = fp + c * D;
        for (int i = 0; i < per; ++i) {
            const int d = lane * per + i;
            a += (float)row[d] * w[d];
        }
#pragma unroll
        for (int o = 16; o > 0; o >>= 1) a += __shfl_xor(a, o, 32);
        s[c] = a;
    }
    float mx = -1e30f;
    for (int c = 0; c < Cdim; ++c) {
        s[c] = 1.0f / (1.0f + __expf(-s[c]));
        mx = fmaxf(mx, s[c]);
    }
    float sum = 0.0f;
    for (int c = 0; c < Cdim; ++c) { s[c] = __expf(s[c] - mx); sum += s[c]; }
    const float inv = 1.0f / sum;

    _Float16* op = out + (size_t)node * out_nstride;
    for (int i = 0; i < per; ++i) {
        const int d = lane * per + i;
        float a = 0.0f;
        for (int c = 0; c < Cdim; ++c) a += s[c] * (float)fp[c * D + d];
        op[d] = (_Float16)(a * inv);
    }
}

// ---------------------------------------------------------------------------
// LayerNorm over [Cdim*HID] per node + PReLU, in-place f16. One block/node.
// ---------------------------------------------------------------------------
__global__ __launch_bounds__(256)
void ln_prelu_kernel(_Float16* __restrict__ x, const float* __restrict__ g,
                     const float* __restrict__ b, const float* __restrict__ alpha,
                     int CH) {
    __shared__ float red[16];
    const int tid  = threadIdx.x;
    const int lane = tid & 31;
    const int wid  = tid >> 5;
    _Float16* xp = x + (size_t)blockIdx.x * CH;

    float sum = 0.0f, sq = 0.0f;
    for (int i = tid; i < CH; i += 256) {
        float v = (float)xp[i];
        sum += v; sq += v * v;
    }
#pragma unroll
    for (int o = 16; o > 0; o >>= 1) {
        sum += __shfl_xor(sum, o, 32);
        sq  += __shfl_xor(sq,  o, 32);
    }
    if (lane == 0) { red[wid] = sum; red[8 + wid] = sq; }
    __syncthreads();
    if (tid == 0) {
        float s = 0.0f, q = 0.0f;
        for (int i = 0; i < 8; ++i) { s += red[i]; q += red[8 + i]; }
        red[0] = s; red[8] = q;
    }
    __syncthreads();
    const float invn = 1.0f / (float)CH;
    const float m  = red[0] * invn;
    const float vr = red[8] * invn - m * m;
    const float rs = rsqrtf(vr + 1e-5f);
    const float a  = *alpha;
    for (int i = tid; i < CH; i += 256) {
        float v = ((float)xp[i] - m) * rs * g[i] + b[i];
        xp[i] = (_Float16)((v >= 0.0f) ? v : a * v);
    }
}

// ---------------------------------------------------------------------------
// Host-side orchestration
// ---------------------------------------------------------------------------
static inline dim3 gemm_grid(int M, int Nn) {
    return dim3((Nn + 127) / 128, M / 128);
}

extern "C" void kernel_launch(void* const* d_in, const int* in_sizes, int n_in,
                              void* d_out, int out_size, void* d_ws, size_t ws_size,
                              hipStream_t stream) {
    constexpr int R = 2, Kf = 3, N = 16384, V = 256;
    constexpr int E = 1, L = 8, VE = 128;
    constexpr int LK = 1, OUT = 64;
    constexpr int IN = 256, HID = 512;
    constexpr int C = Kf + E + LK;  // 5

    const float* raw_feats   = (const float*)d_in[0];
    const float* extra_feats = (const float*)d_in[1];
    const float* label_feats = (const float*)d_in[2];
    const float* feat_emb    = (const float*)d_in[3];
    const float* extra_emb   = (const float*)d_in[4];
    const float* label_emb   = (const float*)d_in[5];
    const float* w_extra     = (const float*)d_in[6];
    const float* w_r         = (const float*)d_in[7];
    const float* w_global    = (const float*)d_in[8];
    const float* W1          = (const float*)d_in[9];
    const float* b1          = (const float*)d_in[10];
    const float* g1          = (const float*)d_in[11];
    const float* be1         = (const float*)d_in[12];
    const float* a1          = (const float*)d_in[13];
    const float* W2          = (const float*)d_in[14];
    const float* b2          = (const float*)d_in[15];
    const float* g2          = (const float*)d_in[16];
    const float* be2         = (const float*)d_in[17];
    const float* a2          = (const float*)d_in[18];
    const float* Wf1         = (const float*)d_in[19];
    const float* bf1         = (const float*)d_in[20];
    const float* af          = (const float*)d_in[21];
    const float* Wf2         = (const float*)d_in[22];
    const float* bf2         = (const float*)d_in[23];
    float* out = (float*)d_out;
    (void)in_sizes; (void)n_in; (void)out_size; (void)ws_size;

    // ---- workspace layout (f16) ----
    char* ws = (char*)d_ws;
    size_t off = 0;
    auto alloc16 = [&](size_t elems) {
        _Float16* p = (_Float16*)(ws + off);
        off += elems * 2;
        return p;
    };
    _Float16* raw16   = alloc16((size_t)R * Kf * N * V);      //  50.3 MB
    _Float16* extra16 = alloc16((size_t)R * E * N * L * VE);  //  67.1 MB
    _Float16* label16 = alloc16((size_t)R * LK * N * OUT);    //   4.2 MB
    _Float16* featT   = alloc16((size_t)R * Kf * IN * V);     //   0.8 MB  [IN,V]
    _Float16* extraT  = alloc16((size_t)R * E * IN * VE);     //   0.1 MB  [IN,VE]
    _Float16* labelT  = alloc16((size_t)R * LK * IN * OUT);   //  0.07 MB  [IN,OUT]
    _Float16* W1T     = alloc16((size_t)C * HID * IN);        //   1.3 MB  [HID,IN]
    _Float16* W2T     = alloc16((size_t)C * HID * HID);       //   2.6 MB
    _Float16* Wf1T    = alloc16((size_t)HID * HID);           //   0.5 MB
    _Float16* Wf2T    = alloc16((size_t)128 * HID);           //  0.13 MB  [64->128 pad, HID]
    _Float16* feats   = alloc16((size_t)N * C * IN);          //  41.9 MB
    _Float16* x1      = alloc16((size_t)N * C * HID);         //  83.9 MB
    _Float16* x2      = alloc16((size_t)N * C * HID);         //  83.9 MB (aliases `me`)
    _Float16* outs    = alloc16((size_t)N * R * HID);         //  33.6 MB
    _Float16* fin     = alloc16((size_t)N * HID);             //  16.8 MB
    _Float16* yb      = alloc16((size_t)N * HID);             //  16.8 MB
    _Float16* me      = x2;  // me [N,L,IN] dead before x2 is written

    const dim3 blk(256);

    // ---- one-time precision/layout pass ----
    {
        size_t n;
        n = (size_t)R * Kf * N * V;
        cvt_f16_kernel<<<(unsigned)(n / 2048), blk, 0, stream>>>(raw_feats, raw16, n);
        n = (size_t)R * E * N * L * VE;
        cvt_f16_kernel<<<(unsigned)(n / 2048), blk, 0, stream>>>(extra_feats, extra16, n);
        n = (size_t)R * LK * N * OUT;
        cvt_f16_kernel<<<(unsigned)(n / 2048), blk, 0, stream>>>(label_feats, label16, n);

        transpose_cvt_kernel<<<dim3(V / 32,  IN / 32,  R * Kf), blk, 0, stream>>>(
            feat_emb,  featT,  V,   IN, (size_t)V * IN,    (size_t)IN * V);
        transpose_cvt_kernel<<<dim3(VE / 32, IN / 32,  R * E),  blk, 0, stream>>>(
            extra_emb, extraT, VE,  IN, (size_t)VE * IN,   (size_t)IN * VE);
        transpose_cvt_kernel<<<dim3(OUT / 32, IN / 32, R * LK), blk, 0, stream>>>(
            label_emb, labelT, OUT, IN, (size_t)OUT * IN,  (size_t)IN * OUT);
        transpose_cvt_kernel<<<dim3(IN / 32,  HID / 32, C),     blk, 0, stream>>>(
            W1,        W1T,    IN,  HID, (size_t)IN * HID, (size_t)HID * IN);
        transpose_cvt_kernel<<<dim3(HID / 32, HID / 32, C),     blk, 0, stream>>>(
            W2,        W2T,    HID, HID, (size_t)HID * HID, (size_t)HID * HID);
        transpose_cvt_kernel<<<dim3(HID / 32, HID / 32, 1),     blk, 0, stream>>>(
            Wf1,       Wf1T,   HID, HID, (size_t)HID * HID, (size_t)HID * HID);
        transpose_cvt_kernel<<<dim3(HID / 32, 128 / 32, 1),     blk, 0, stream>>>(
            Wf2,       Wf2T,   HID, OUT, (size_t)HID * OUT, (size_t)128 * HID);
    }

    for (int r = 0; r < R; ++r) {
        // mapped: per k, [N,V] x Bt[IN,V] -> feats[:, k, :]
        for (int k = 0; k < Kf; ++k) {
            gemm_wmma_async<_Float16><<<gemm_grid(N, IN), blk, 0, stream>>>(
                raw16 + (size_t)(r * Kf + k) * N * V, V,
                featT + (size_t)(r * Kf + k) * IN * V,
                feats + k * IN, C * IN, N, IN, V, nullptr, nullptr);
        }
        // extra: [N*L, VE] x Bt[IN,VE] -> me[N,L,IN]
        gemm_wmma_async<_Float16><<<gemm_grid(N * L, IN), blk, 0, stream>>>(
            extra16 + (size_t)r * E * N * L * VE, VE,
            extraT  + (size_t)r * E * IN * VE,
            me, IN, N * L, IN, VE, nullptr, nullptr);
        // attn pool over L -> feats[:, Kf, :]
        attn_pool_kernel<<<N / 8, blk, 0, stream>>>(
            me, w_extra, feats + Kf * IN, N, L, IN, L * IN, C * IN);
        // label: [N,OUT] x Bt[IN,OUT] -> feats[:, Kf+E, :]
        gemm_wmma_async<_Float16><<<gemm_grid(N, IN), blk, 0, stream>>>(
            label16 + (size_t)r * LK * N * OUT, OUT,
            labelT  + (size_t)r * LK * IN * OUT,
            feats + (Kf + E) * IN, C * IN, N, IN, OUT, nullptr, nullptr);

        // grouped conv1: per c, [N,IN] x Bt[HID,IN] + b1[c] -> x1[:, c, :]
        for (int c = 0; c < C; ++c) {
            gemm_wmma_async<_Float16><<<gemm_grid(N, HID), blk, 0, stream>>>(
                feats + c * IN, C * IN,
                W1T + (size_t)c * HID * IN,
                x1 + c * HID, C * HID, N, HID, IN, b1 + c * HID, nullptr);
        }
        ln_prelu_kernel<<<N, blk, 0, stream>>>(x1, g1, be1, a1, C * HID);

        // grouped conv2: per c, [N,HID] x Bt[HID,HID] + b2[c] -> x2[:, c, :]
        for (int c = 0; c < C; ++c) {
            gemm_wmma_async<_Float16><<<gemm_grid(N, HID), blk, 0, stream>>>(
                x1 + c * HID, C * HID,
                W2T + (size_t)c * HID * HID,
                x2 + c * HID, C * HID, N, HID, HID, b2 + c * HID, nullptr);
        }
        ln_prelu_kernel<<<N, blk, 0, stream>>>(x2, g2, be2, a2, C * HID);

        // attn pool over C with w_r[r] -> outs[:, r, :]
        attn_pool_kernel<<<N / 8, blk, 0, stream>>>(
            x2, w_r + (size_t)r * HID, outs + r * HID, N, C, HID, C * HID, R * HID);
    }

    // global attn pool over R -> fin
    attn_pool_kernel<<<N / 8, blk, 0, stream>>>(
        outs, w_global, fin, N, R, HID, R * HID, HID);

    // FFN
    gemm_wmma_async<_Float16><<<gemm_grid(N, HID), blk, 0, stream>>>(
        fin, HID, Wf1T, yb, HID, N, HID, HID, bf1, af);
    gemm_wmma_async<float><<<gemm_grid(N, OUT), blk, 0, stream>>>(
        yb, HID, Wf2T, out, OUT, N, OUT, HID, bf2, nullptr);
}
